// TradingModel_8701603741750
// MI455X (gfx1250) — compile-verified, matrix operand order checked by reading, not run on hardware
//
#include <hip/hip_runtime.h>
#include <hip/hip_bf16.h>
#include <math.h>

#define LOOKBACK  512
#define T_LEN     131072
#define N_STEPS   (T_LEN - LOOKBACK + 1)      // 130561
#define N_TILES   ((N_STEPS + 255) / 256)     // 511 tiles of 256 outputs
#define KPAD      544                         // 512 + 15 shifted, padded to 17*32
#define KBLK      (KPAD / 32)                 // 17 WMMA K-blocks
#define FEAT_PAD  (T_LEN + 512)               // zero-padded f16 features
#define DOT_PAD   (N_TILES * 256)             // 130816
#define FEES_F    0.00025f

typedef _Float16 half8 __attribute__((ext_vector_type(8)));
typedef _Float16 v16h  __attribute__((ext_vector_type(16)));
typedef float    v8f   __attribute__((ext_vector_type(8)));

// ---------------------------------------------------------------------------
// 1) Convert features f32 -> f16, zero-padded to FEAT_PAD
// ---------------------------------------------------------------------------
__global__ void cvt_features_kernel(const float* __restrict__ feat,
                                    _Float16* __restrict__ f16feat) {
    int i = blockIdx.x * blockDim.x + threadIdx.x;
    if (i < FEAT_PAD) {
        float v = (i < T_LEN) ? feat[i] : 0.0f;
        f16feat[i] = (_Float16)v;
    }
}

// ---------------------------------------------------------------------------
// 2) Build Toeplitz B-matrix (B[k][j] = theta[k-j], 0 outside [0,512)) directly
//    in the wave32 WMMA B-fragment layout: bfrag[kb][lane][q], 16 halves/lane.
//    Lane L: N = L&15; q<8 -> K = kb*32 + base + q, q>=8 -> K = kb*32+base+16+(q-8)
//    where base = (L<16) ? 0 : 8.
// ---------------------------------------------------------------------------
__global__ void build_bfrag_kernel(const float* __restrict__ theta,
                                   _Float16* __restrict__ bfrag) {
    int idx = blockIdx.x * blockDim.x + threadIdx.x;   // [0, KBLK*32*16)
    if (idx >= KBLK * 32 * 16) return;
    int kb   = idx >> 9;          // /512
    int rem  = idx & 511;
    int lane = rem >> 4;
    int q    = rem & 15;
    int N    = lane & 15;
    int base = (lane < 16) ? 0 : 8;
    int K    = kb * 32 + ((q < 8) ? (base + q) : (base + 16 + (q - 8)));
    int m    = K - N;
    float v  = (m >= 0 && m < LOOKBACK) ? theta[m] : 0.0f;
    bfrag[idx] = (_Float16)v;
}

// ---------------------------------------------------------------------------
// 3) Per-step window sum and sum-of-squares (features fit in L2: 512 KB)
// ---------------------------------------------------------------------------
__global__ void winstats_kernel(const float* __restrict__ feat,
                                float* __restrict__ wsum,
                                float* __restrict__ wss) {
    int t = blockIdx.x * blockDim.x + threadIdx.x;
    if (t >= N_STEPS) return;
    float s = 0.0f, ss = 0.0f;
    #pragma unroll 8
    for (int k = 0; k < LOOKBACK; ++k) {
        float v = feat[t + k];
        s  += v;
        ss += v * v;
    }
    wsum[t] = s;
    wss[t]  = ss;
}

// ---------------------------------------------------------------------------
// 4) Sliding dot products S[t] = sum_k f[t+k]*theta[k] as WMMA GEMM.
//    One wave per tile of 256 outputs. C[M][N] = S[t0 + 16*M + N].
// ---------------------------------------------------------------------------
__global__ __launch_bounds__(32) void conv_wmma_kernel(
        const _Float16* __restrict__ f16feat,
        const _Float16* __restrict__ bfrag,
        float* __restrict__ dot) {
    const int t0   = blockIdx.x * 256;
    const int lane = threadIdx.x;                 // 0..31, wave32
    const int M    = lane & 15;
    const int base = (lane < 16) ? 0 : 8;

    const _Float16* arow = f16feat + t0 + 16 * M; // contiguous row for this lane

    v8f c = {};
    #pragma unroll
    for (int kb = 0; kb < KBLK; ++kb) {
        const int k0 = kb * 32;
        // A fragment: two 16B contiguous loads per lane
        half8 alo = *(const half8*)(arow + k0 + base);
        half8 ahi = *(const half8*)(arow + k0 + base + 16);
        // B fragment: prebuilt, 32B contiguous per lane
        v16h b = *(const v16h*)(bfrag + (kb * 32 + lane) * 16);
        v16h a;
        #pragma unroll
        for (int q = 0; q < 8; ++q) { a[q] = alo[q]; a[q + 8] = ahi[q]; }
        c = __builtin_amdgcn_wmma_f32_16x16x32_f16(
                /*neg_a=*/false, a, /*neg_b=*/false, b,
                /*c_mod=*/(short)0, c, /*reuse_a=*/false, /*reuse_b=*/false);
    }

    // C/D layout: lane<16 -> rows 0..7, lane>=16 -> rows 8..15; N = lane&15
    const int moff = (lane < 16) ? 0 : 8;
    const int N    = lane & 15;
    #pragma unroll
    for (int r = 0; r < 8; ++r) {
        int t = t0 + 16 * (r + moff) + N;
        if (t < N_STEPS) dot[t] = c[r];
    }
}

// ---------------------------------------------------------------------------
// 5) Sequential positions scan (latency-bound nonlinear recurrence).
//    Uses exact identities: sum(win_norm)=0, sum(win_norm^2)=LOOKBACK-1.
// ---------------------------------------------------------------------------
__global__ void positions_kernel(const float* __restrict__ theta,
                                 const float* __restrict__ p_init,
                                 const float* __restrict__ dot,
                                 const float* __restrict__ wsum,
                                 const float* __restrict__ wss,
                                 float* __restrict__ P) {
    if (threadIdx.x != 0 || blockIdx.x != 0) return;

    float sum512 = 0.0f;
    for (int k = 0; k < LOOKBACK; ++k) sum512 += theta[k];
    const float th512  = theta[LOOKBACK];
    const float th513  = theta[LOOKBACK + 1];
    const float sum514 = sum512 + th512 + th513;

    float n = (float)(LOOKBACK - 1);
    float s = 0.0f, ss = 0.0f;
    for (int i = 0; i < LOOKBACK - 1; ++i) {
        float p = p_init[i];
        s += p; ss += p * p;
        P[i] = p;
    }
    float p_last = p_init[LOOKBACK - 2];

    const float LBf = (float)LOOKBACK;            // 512
    const float DIM = (float)(LOOKBACK + 2);      // 514

    for (int t = 0; t < N_STEPS; ++t) {
        // running stats of positions so far
        float mean = s / n;
        float stdv = sqrtf((ss - n * mean * mean) / (n - 1.0f));
        float ln   = (p_last - mean) / stdv;
        // window stats
        float wm   = wsum[t] * (1.0f / LBf);
        float wstd = sqrtf((wss[t] - LBf * wm * wm) * (1.0f / (LBf - 1.0f)));
        // theta[0:512] . win_norm  (hoisted GEMM result)
        float d    = (dot[t] - wm * sum512) / wstd;
        // xt re-normalization via exact identities
        float dotx = d + th512 * ln + th513;
        float mx   = (ln + 1.0f) * (1.0f / DIM);
        float sq   = (LBf - 1.0f) + ln * ln + 1.0f;           // sum(xt^2)
        float sx   = sqrtf((sq - DIM * mx * mx) * (1.0f / (DIM - 1.0f)));
        float z    = (dotx - mx * sum514) / sx;
        float p    = 1.0f / (1.0f + expf(-z));
        P[(LOOKBACK - 1) + t] = p;
        n += 1.0f; s += p; ss += p * p; p_last = p;
    }
}

// ---------------------------------------------------------------------------
// 6) Sequential portfolio scan -> per-step return ratios (output, length T)
// ---------------------------------------------------------------------------
__global__ void botret_kernel(const float* __restrict__ P,
                              const float* __restrict__ ar,
                              float* __restrict__ out) {
    if (threadIdx.x != 0 || blockIdx.x != 0) return;
    float inv  = P[0];
    float cash = 1.0f - inv;
    float pv_prev = 1.0f;
    out[0] = 1.0f;
    for (int t = 1; t < T_LEN; ++t) {
        float r = ar[t];
        float p = P[t];
        inv *= r;
        float pv    = inv + cash;
        float delta = pv * p - inv;
        if (delta > 0.0f) { inv += delta * (1.0f - FEES_F); cash -= delta; }
        else              { inv += delta; cash -= delta * (1.0f - FEES_F); }
        float pvn = inv + cash;
        out[t] = pvn / pv_prev;
        pv_prev = pvn;
    }
}

// ---------------------------------------------------------------------------
extern "C" void kernel_launch(void* const* d_in, const int* in_sizes, int n_in,
                              void* d_out, int out_size, void* d_ws, size_t ws_size,
                              hipStream_t stream) {
    const float* features      = (const float*)d_in[0];
    const float* asset_returns = (const float*)d_in[1];
    const float* theta         = (const float*)d_in[2];
    const float* p_init        = (const float*)d_in[3];
    float* out = (float*)d_out;

    // workspace layout (256B-aligned sections)
    char* ws = (char*)d_ws;
    size_t off = 0;
    _Float16* f16feat = (_Float16*)(ws + off); off += (size_t)FEAT_PAD * 2;         // 263168
    _Float16* bfrag   = (_Float16*)(ws + off); off += (size_t)KBLK * 32 * 16 * 2;   // 17408
    float*    dot     = (float*)(ws + off);    off += (size_t)DOT_PAD * 4;          // 523264
    float*    wsum    = (float*)(ws + off);    off += (size_t)DOT_PAD * 4;
    float*    wss     = (float*)(ws + off);    off += (size_t)DOT_PAD * 4;
    float*    P       = (float*)(ws + off);    off += (size_t)T_LEN * 4;

    cvt_features_kernel<<<(FEAT_PAD + 255) / 256, 256, 0, stream>>>(features, f16feat);
    build_bfrag_kernel<<<(KBLK * 32 * 16 + 255) / 256, 256, 0, stream>>>(theta, bfrag);
    winstats_kernel<<<(N_STEPS + 255) / 256, 256, 0, stream>>>(features, wsum, wss);
    conv_wmma_kernel<<<N_TILES, 32, 0, stream>>>(f16feat, bfrag, dot);
    positions_kernel<<<1, 1, 0, stream>>>(theta, p_init, dot, wsum, wss, P);
    botret_kernel<<<1, 1, 0, stream>>>(P, asset_returns, out);
}
